// SpatialCirConv_36051955482654
// MI455X (gfx1250) — matrix operand order repaired and checked by reading, not compiled, not run on hardware
//
#include <hip/hip_runtime.h>
#include <math.h>
#include <stdint.h>

// ---------------------------------------------------------------------------
// SpatialCirConv on MI455X (gfx1250).
//
// rfft_block( conv3d( irfft_block(X), circulant(W) ) ) ==
//   per-frequency complex conv:  Y_hat[k] = fft_block(W)[k] (*) X_hat[k]
// Complex channels as real 2x2 block => per frequency k (5 bins):
//   GEMM M=64 (32 re + 32 im out), K = 32 (16 re + 16 im in) x 27 taps,
//   N = 4*16*56*56 = 200704 positions.  bf16 WMMA 16x16x32, fp32 accumulate.
//
// v3: weight stage via Tensor Data Mover (tensor_load_to_lds + TENSORcnt)
//     instead of a per-thread load/ds_store loop.
// ---------------------------------------------------------------------------

typedef __attribute__((ext_vector_type(16))) __bf16 v16bf;
typedef __attribute__((ext_vector_type(8)))  float  v8f;
typedef __attribute__((ext_vector_type(4)))  unsigned int u32x4;
typedef __attribute__((ext_vector_type(8)))  int i32x8;
typedef __attribute__((ext_vector_type(4)))  int i32x4;

union ABFrag { v16bf v; u32x4 q[2]; };

#define NB 8        // circulant block size
#define NF 5        // rfft bins = NB/2+1
#define INB 16
#define ONB 32
#define TAPS 27
#define DHW 50176   // 16*56*56
#define NPOS 200704 // 4 * DHW
#define NTILE 1568  // NPOS / 128  (workgroup N tile = 128)
#define MROWS 64    // 2*ONB (re/im block rows)
#define KCH 32      // 2*INB (re/im block cols)
// padded activation volume per (k,b): 18 x 58 x 58
#define PZ 18
#define PY 58
#define PX 58
#define PVOL (PZ*PY*PX)   // 60552
#define WTILE_B8 (TAPS*MROWS*KCH/4)   // weight image in 8-byte units = 13824

__device__ __forceinline__ uint16_t f2bf(float f) {
    uint32_t u = __builtin_bit_cast(uint32_t, f);
    uint32_t r = u + 0x7FFFu + ((u >> 16) & 1u);   // round-to-nearest-even
    return (uint16_t)(r >> 16);
}

// --- Kernel 1: frequency-domain block weights -------------------------------
// Wf[k][tap][m][c] bf16,  m<32: [Wr | -Wi], m>=32: [Wi | Wr]
__global__ void prep_w_kernel(const float* __restrict__ w, uint16_t* __restrict__ Wf) {
    int idx = blockIdx.x * 256 + threadIdx.x;
    if (idx >= NF * TAPS * MROWS * KCH) return;
    int c   = idx & (KCH - 1);
    int t1  = idx >> 5;
    int m   = t1 & (MROWS - 1);
    int t2  = t1 >> 6;
    int tap = t2 % TAPS;
    int k   = t2 / TAPS;
    int ri  = m >> 5, onb = m & 31;
    int ci  = c >> 4, inb = c & 15;
    float wr = 0.f, wi = 0.f;
    for (int d = 0; d < NB; ++d) {
        float ang = -2.0f * (float)M_PI * (float)(k * d) / (float)NB;
        float val = w[((onb * INB + inb) * NB + d) * TAPS + tap];
        wr += val * cosf(ang);
        wi += val * sinf(ang);
    }
    float v = (ri == 0) ? (ci == 0 ? wr : -wi) : (ci == 0 ? wi : wr);
    Wf[idx] = f2bf(v);
}

// --- Kernel 2: bias DFT along block axis ------------------------------------
__global__ void prep_bias_kernel(const float* __restrict__ bias, float* __restrict__ bhat) {
    int idx = threadIdx.x;
    if (idx >= NF * ONB) return;
    int onb = idx % ONB, k = idx / ONB;
    float re = 0.f, im = 0.f;
    for (int j = 0; j < NB; ++j) {
        float ang = -2.0f * (float)M_PI * (float)(k * j) / (float)NB;
        float v = bias[onb * NB + j];
        re += v * cosf(ang);
        im += v * sinf(ang);
    }
    bhat[idx * 2 + 0] = re;
    bhat[idx * 2 + 1] = im;
}

// --- Kernel 3: pack X into zero-padded bf16 NDHWC frequency tiles -----------
// Xpad[k][b][zz(18)][yy(58)][xx(58)][32]: ch<16 real(inb), ch>=16 imag(inb).
__global__ void pack_x_kernel(const float* __restrict__ xr, const float* __restrict__ xi,
                              uint16_t* __restrict__ Xpad) {
    long idx = (long)blockIdx.x * 256 + threadIdx.x;       // (k*4+b)*PVOL + p
    if (idx >= (long)NF * 4 * PVOL) return;
    int p  = (int)(idx % PVOL);
    int kb = (int)(idx / PVOL);
    int b  = kb & 3, k = kb >> 2;
    int xx = p % PX;
    int yy = (p / PX) % PY;
    int zz = p / (PX * PY);

    uint16_t o[KCH];
    bool interior = (zz >= 1) & (zz <= 16) & (yy >= 1) & (yy <= 56) &
                    (xx >= 1) & (xx <= 56);
    if (interior) {
        int s = (zz - 1) * 3136 + (yy - 1) * 56 + (xx - 1);
        for (int inb = 0; inb < INB; ++inb) {
            long src = ((long)b * (INB * NF) + inb * NF + k) * DHW + s;
            o[inb]       = f2bf(xr[src]);
            o[INB + inb] = f2bf(xi[src]);
        }
    } else {
        for (int c = 0; c < KCH; ++c) o[c] = 0;
    }
    u32x4* dst = (u32x4*)(Xpad + idx * KCH);
    const u32x4* srcv = (const u32x4*)o;
    dst[0] = srcv[0]; dst[1] = srcv[1]; dst[2] = srcv[2]; dst[3] = srcv[3];
}

// --- Kernel 4: implicit-GEMM conv via bf16 WMMA -----------------------------
// grid.x = NF * NTILE ; block = 256 (8 wave32s).
// Workgroup tile M=64 x N=128; each wave owns an N=16 strip, loops 4 M-tiles.
// Weight image (110 KB) DMA'd to LDS once per workgroup by the TDM.
__global__ __launch_bounds__(256) void conv_wmma_kernel(
    const uint16_t* __restrict__ Wf, const uint16_t* __restrict__ Xpad,
    const float* __restrict__ bhat, float* __restrict__ out) {

    __shared__ __align__(16) uint16_t sW[TAPS * MROWS * KCH];  // 110,592 B in 320 KB LDS

    const int tid  = threadIdx.x;
    const int k    = blockIdx.x / NTILE;
    const int tile = blockIdx.x % NTILE;

    // ---- TDM weight stage: one 1-D descriptor, issued by wave 0 -----------
    if (tid < 32) {
        // lds byte address = low 32 bits of generic shared address
        unsigned lds_addr = (unsigned)(uintptr_t)(void*)sW;
        unsigned long long ga =
            (unsigned long long)(uintptr_t)(Wf + (size_t)k * TAPS * MROWS * KCH);
        u32x4 g0;
        g0[0] = 1u;                                        // count=1, user mode
        g0[1] = lds_addr;                                  // D#.lds_addr
        g0[2] = (unsigned)(ga & 0xFFFFFFFFu);              // global_addr[31:0]
        g0[3] = (unsigned)((ga >> 32) & 0x01FFFFFFu)       // global_addr[56:32]
              | (2u << 30);                                // type = 2 ("image")
        i32x8 g1;
        g1[0] = (3 << 16);                                 // data_size = 8 bytes
        g1[1] = (int)((WTILE_B8 & 0xFFFFu) << 16);         // tensor_dim0[15:0]
        g1[2] = (int)(WTILE_B8 >> 16);                     // tensor_dim0[31:16]
        g1[3] = (int)((unsigned)WTILE_B8 << 16);           // tile_dim0 = 13824
        g1[4] = 0;                                         // tile_dim1/2 = 0 (1-D)
        g1[5] = (int)WTILE_B8;                             // tensor_dim0_stride
        g1[6] = 0;
        g1[7] = 0;
        i32x4 gz = {0, 0, 0, 0};
#if __clang_major__ >= 23
        i32x8 gz8 = {0, 0, 0, 0, 0, 0, 0, 0};
        __builtin_amdgcn_tensor_load_to_lds(g0, g1, gz, gz, gz8, 0);
#else
        __builtin_amdgcn_tensor_load_to_lds(g0, g1, gz, gz, 0);
#endif
        __builtin_amdgcn_s_wait_tensorcnt(0);
    }
    __syncthreads();

    const int wave = tid >> 5;          // wave32
    const int lane = tid & 31;
    const int l16  = lane & 15;
    const int s0   = lane >> 4;         // half-wave select (ISA 16-bit A/B layouts)
    const int n0   = wave * 16;         // this wave's N strip

    // this lane's spatial column (NPOS % 128 == 0, always in range)
    const int pos = tile * 128 + n0 + l16;
    const int b = pos / DHW;
    const int s = pos % DHW;
    const int z = s / 3136;
    const int y = (s / 56) % 56;
    const int x = s % 56;

    // center address in padded volume; tap deltas become constant IOFFSETs
    const uint16_t* bbase = Xpad +
        ((((size_t)(k * 4 + b) * PZ + (z + 1)) * PY + (y + 1)) * PX + (x + 1)) * KCH
        + 8 * s0;
    // A fragment row base in LDS (elements [8s0..+7] and [16+8s0..+7] of row)
    const uint16_t* abase = sW + l16 * KCH + 8 * s0;

    v8f acc[4] = {};

#pragma unroll
    for (int tap = 0; tap < TAPS; ++tap) {
        const int dz = tap / 9 - 1;
        const int dy = (tap / 3) % 3 - 1;
        const int dx = tap % 3 - 1;

        ABFrag bf;
        const u32x4* bp = (const u32x4*)(bbase + (dz * (PY * PX) + dy * PX + dx) * KCH);
        bf.q[0] = bp[0];    // channels 8*s0 .. +7
        bf.q[1] = bp[2];    // channels 16+8*s0 .. +7

#pragma unroll
        for (int mt = 0; mt < 4; ++mt) {
            ABFrag a;
            const u32x4* ap = (const u32x4*)(abase + tap * (MROWS * KCH) + mt * (16 * KCH));
            a.q[0] = ap[0];   // channels  8*s0 .. +7
            a.q[1] = ap[2];   // channels 16+8*s0 .. +7
            acc[mt] = __builtin_amdgcn_wmma_f32_16x16x32_bf16(
                false, a.v, false, bf.v, (short)0, acc[mt], false, false);
        }
    }

    // epilogue: D layout VGPR r -> (M = mt*16 + r + 8*s0, N = n0 + l16)
#pragma unroll
    for (int mt = 0; mt < 4; ++mt) {
#pragma unroll
        for (int r = 0; r < 8; ++r) {
            int m   = mt * 16 + r + 8 * s0;
            int ri  = m >> 5;           // 0 = real part, 1 = imag part
            int onb = m & 31;
            float v = acc[mt][r] + bhat[(k * ONB + onb) * 2 + ri];
            size_t oidx = (((size_t)ri * 4 + b) * (ONB * NF) + (onb * NF + k)) * DHW + s;
            out[oidx] = v;
        }
    }
}

// ---------------------------------------------------------------------------
extern "C" void kernel_launch(void* const* d_in, const int* in_sizes, int n_in,
                              void* d_out, int out_size, void* d_ws, size_t ws_size,
                              hipStream_t stream) {
    (void)in_sizes; (void)n_in; (void)out_size; (void)ws_size;
    const float* xr   = (const float*)d_in[0];
    const float* xi   = (const float*)d_in[1];
    const float* w    = (const float*)d_in[2];
    const float* bias = (const float*)d_in[3];
    float* out = (float*)d_out;

    // workspace layout
    char* ws = (char*)d_ws;
    constexpr size_t WF_BYTES   = (size_t)NF * TAPS * MROWS * KCH * 2;  // 552,960
    constexpr size_t BHAT_BYTES = (size_t)NF * ONB * 2 * 4;             //   1,280
    uint16_t* Wf   = (uint16_t*)ws;
    float*    bhat = (float*)(ws + WF_BYTES);
    uint16_t* Xpad = (uint16_t*)(ws + WF_BYTES + BHAT_BYTES);           // ~77.5 MB

    int nW = NF * TAPS * MROWS * KCH;
    prep_w_kernel<<<(nW + 255) / 256, 256, 0, stream>>>(w, Wf);
    prep_bias_kernel<<<1, 256, 0, stream>>>(bias, bhat);

    long nX = (long)NF * 4 * PVOL;
    pack_x_kernel<<<(unsigned)((nX + 255) / 256), 256, 0, stream>>>(xr, xi, Xpad);

    conv_wmma_kernel<<<NF * NTILE, 256, 0, stream>>>(Wf, Xpad, bhat, out);
}
